// LiquidStateMachine_49658411876579
// MI455X (gfx1250) — matrix operand order repaired and checked by reading, not compiled, not run on hardware
//
#include <hip/hip_runtime.h>
#include <hip/hip_bf16.h>
#include <stdint.h>

// LSM / Echo-State forward:  h = tanh(x@Wx + h@Wh); y = h@Wro + b
// B=64, T=2048, IN=128, LIQ=512, OUT=128.
// 4 persistent workgroups (one per M=16 batch tile), 8 waves each.
// - [Wx;Wh] pre-packed to bf16 WMMA B-fragments, streamed from L2 each step.
// - W_ro fragments held in VGPRs for the whole T loop (128 regs/lane).
// - h lives in LDS; tanh via hardware V_TANH_F32.
// - x_{t+1} prefetched into registers during step t (HBM latency off the chain).
// - Readout uses two independent WMMA accumulator chains to overlap LDS loads.

typedef __attribute__((ext_vector_type(16))) __bf16 v16bf;
typedef __attribute__((ext_vector_type(8)))  float  v8f;
typedef __attribute__((ext_vector_type(4)))  float  v4f;   // clang vector: NT-load legal

#define B_    64
#define T_    2048
#define IN_   128
#define LIQ_  512
#define OUT_  128
#define KTOT  (IN_ + LIQ_)     // 640 fused K for [x|h] @ [Wx;Wh]
#define KB_W  (KTOT / 32)      // 20 k-steps, main GEMM
#define KB_RO (LIQ_ / 32)      // 16 k-steps, readout GEMM
#define LDS_STRIDE 648         // bf16 elems/row: 1296 B, 16B-aligned, bank-spread

static __device__ __forceinline__ uint32_t f32_to_bf16_rne(float f) {
    uint32_t u = __float_as_uint(f);
    return (u + 0x7FFFu + ((u >> 16) & 1u)) >> 16;
}

static __device__ __forceinline__ float fast_tanh(float x) {
#if __has_builtin(__builtin_amdgcn_tanhf)
    return __builtin_amdgcn_tanhf(x);   // V_TANH_F32 (CDNA5 TRANS op)
#else
    return tanhf(x);
#endif
}

// Pack a row-major KxN fp32 matrix into bf16 WMMA B-fragments.
// Fragment dword j (0..7) of lane l for tile (nt,kb):
//   col = nt*16 + (l&15);  k = kb*32 + (l>>4)*8 + (j>>2)*8 + 2j  ->  {k, k+1} packed lo/hi.
// Tiles stored nt-major (tileIdx = nt*KB + kb) so each wave streams contiguously.
__global__ void lsm_pack_b(const float* __restrict__ src, int N, int KB,
                           uint32_t* __restrict__ dst, int totalDwords) {
    int idx = blockIdx.x * blockDim.x + threadIdx.x;
    if (idx >= totalDwords) return;
    int j    = idx & 7;
    int lane = (idx >> 3) & 31;
    int tile = idx >> 8;
    int nt   = tile / KB;
    int kb   = tile - nt * KB;
    int col  = nt * 16 + (lane & 15);
    int k    = kb * 32 + (lane >> 4) * 8 + (j >> 2) * 8 + 2 * j;
    uint32_t lo = f32_to_bf16_rne(src[(size_t)k * N + col]);
    uint32_t hi = f32_to_bf16_rne(src[(size_t)(k + 1) * N + col]);
    dst[idx] = (hi << 16) | lo;
}

__global__ __launch_bounds__(256, 1)
void lsm_main(const float* __restrict__ x,     // [B,T,IN] fp32
              const uint4* __restrict__ wxh,   // packed [Wx;Wh] bf16 frags
              const uint4* __restrict__ wro,   // packed Wro bf16 frags
              const float* __restrict__ b_ro,  // [OUT]
              float* __restrict__ y_out,       // [B,T,OUT]
              float* __restrict__ h_out)       // [B,LIQ]
{
    __shared__ __align__(16) unsigned short xh[16 * LDS_STRIDE]; // [x_t | h] bf16

    const int tid  = threadIdx.x;
    const int wave = tid >> 5;
    const int lane = tid & 31;
    const int hsel = lane >> 4;      // half-wave select (K split for A/B, row split for C)
    const int lcol = lane & 15;
    const int b0   = blockIdx.x * 16;

    union Frag { uint4 u[2]; v16bf v; };

    // h := 0
    {
        int r  = tid >> 4;
        int c0 = IN_ + (tid & 15) * 32;
        uint4 z = make_uint4(0u, 0u, 0u, 0u);
        uint4* p = (uint4*)&xh[r * LDS_STRIDE + c0];
        p[0] = z; p[1] = z; p[2] = z; p[3] = z;
    }

    const float bias = b_ro[wave * 16 + lcol];

    // W_ro fragments: resident in VGPRs for the whole sequence (16 x 8 dwords/lane)
    v16bf wro_f[KB_RO];
    #pragma unroll
    for (int kb = 0; kb < KB_RO; ++kb) {
        const uint4* bp = wro + ((size_t)wave * KB_RO + kb) * 64 + lane * 2;
        Frag f;
        f.u[0] = bp[0];
        f.u[1] = bp[1];
        wro_f[kb] = f.v;
    }

    // per-thread x staging geometry + software-pipelined prefetch registers
    const int xr  = tid >> 4;             // row 0..15
    const int xc0 = (tid & 15) * 8;       // col 0..120
    const float* xptr = x + ((size_t)(b0 + xr) * T_) * IN_ + xc0;
    v4f xf0, xf1;                         // x_t values in flight
    xf0 = __builtin_nontemporal_load((const v4f*)(xptr + 0));
    xf1 = __builtin_nontemporal_load((const v4f*)(xptr + 4));

    for (int t = 0; t < T_; ++t) {
        __syncthreads();
        // stage prefetched x_t -> LDS as bf16
        {
            uint32_t d0 = (f32_to_bf16_rne(xf0.y) << 16) | f32_to_bf16_rne(xf0.x);
            uint32_t d1 = (f32_to_bf16_rne(xf0.w) << 16) | f32_to_bf16_rne(xf0.z);
            uint32_t d2 = (f32_to_bf16_rne(xf1.y) << 16) | f32_to_bf16_rne(xf1.x);
            uint32_t d3 = (f32_to_bf16_rne(xf1.w) << 16) | f32_to_bf16_rne(xf1.z);
            *(uint4*)&xh[xr * LDS_STRIDE + xc0] = make_uint4(d0, d1, d2, d3);
        }
        // prefetch x_{t+1}: latency hidden under this whole step
        if (t + 1 < T_) {
            const float* xn = xptr + (size_t)(t + 1) * IN_;
            xf0 = __builtin_nontemporal_load((const v4f*)(xn + 0));
            xf1 = __builtin_nontemporal_load((const v4f*)(xn + 4));
        }
        __syncthreads();

        // main GEMM: acc(16x64 per wave) = [x|h](16x640) @ [Wx;Wh](640x64 slice)
        v8f acc0 = {}, acc1 = {}, acc2 = {}, acc3 = {};
        #pragma unroll 4
        for (int kb = 0; kb < KB_W; ++kb) {
            Frag a;
            const uint4* ap = (const uint4*)&xh[lcol * LDS_STRIDE + kb * 32 + hsel * 8];
            a.u[0] = ap[0];      // K = b .. b+7
            a.u[1] = ap[2];      // K = b+16 .. b+23
            const uint4* bp = wxh + ((size_t)(wave * 4) * KB_W + kb) * 64 + lane * 2;
            Frag b;
            b.u[0] = bp[0]; b.u[1] = bp[1];
            acc0 = __builtin_amdgcn_wmma_f32_16x16x32_bf16(false, a.v, false, b.v, (short)0, acc0, false, false);
            bp += (size_t)KB_W * 64;
            b.u[0] = bp[0]; b.u[1] = bp[1];
            acc1 = __builtin_amdgcn_wmma_f32_16x16x32_bf16(false, a.v, false, b.v, (short)0, acc1, false, false);
            bp += (size_t)KB_W * 64;
            b.u[0] = bp[0]; b.u[1] = bp[1];
            acc2 = __builtin_amdgcn_wmma_f32_16x16x32_bf16(false, a.v, false, b.v, (short)0, acc2, false, false);
            bp += (size_t)KB_W * 64;
            b.u[0] = bp[0]; b.u[1] = bp[1];
            acc3 = __builtin_amdgcn_wmma_f32_16x16x32_bf16(false, a.v, false, b.v, (short)0, acc3, false, false);
        }
        __syncthreads();   // all waves done reading old h

        // h = tanh(acc); write bf16 back into LDS (C layout: VGPR e -> M=e+8*hsel, N=lcol)
        #pragma unroll
        for (int i = 0; i < 4; ++i) {
            v8f a = (i == 0) ? acc0 : (i == 1) ? acc1 : (i == 2) ? acc2 : acc3;
            int n = (wave * 4 + i) * 16 + lcol;
            #pragma unroll
            for (int e = 0; e < 8; ++e) {
                float v = fast_tanh(a[e]);
                int m = e + hsel * 8;
                xh[m * LDS_STRIDE + IN_ + n] = (unsigned short)f32_to_bf16_rne(v);
                if (t == T_ - 1)
                    h_out[(size_t)(b0 + m) * LIQ_ + n] = v;
            }
        }
        __syncthreads();

        // readout: y(16x16 per wave) = h(16x512) @ Wro(512x16 slice) + b
        // Two independent accumulator chains; both A-frag loads issued before
        // either WMMA so LDS latency hides under the other chain's WMMA.
        v8f yacc0 = {bias, bias, bias, bias, bias, bias, bias, bias};
        v8f yacc1 = {};
        #pragma unroll
        for (int kb = 0; kb < KB_RO; kb += 2) {
            Frag a0, a1;
            const uint4* ap0 = (const uint4*)&xh[lcol * LDS_STRIDE + IN_ + kb * 32 + hsel * 8];
            const uint4* ap1 = (const uint4*)&xh[lcol * LDS_STRIDE + IN_ + (kb + 1) * 32 + hsel * 8];
            a0.u[0] = ap0[0]; a0.u[1] = ap0[2];
            a1.u[0] = ap1[0]; a1.u[1] = ap1[2];
            yacc0 = __builtin_amdgcn_wmma_f32_16x16x32_bf16(false, a0.v, false, wro_f[kb],     (short)0, yacc0, false, false);
            yacc1 = __builtin_amdgcn_wmma_f32_16x16x32_bf16(false, a1.v, false, wro_f[kb + 1], (short)0, yacc1, false, false);
        }
        v8f yacc = yacc0 + yacc1;
        {
            int n = wave * 16 + lcol;
            #pragma unroll
            for (int e = 0; e < 8; ++e) {
                int m = e + hsel * 8;
                __builtin_nontemporal_store(yacc[e],
                    y_out + ((size_t)(b0 + m) * T_ + t) * OUT_ + n);
            }
        }
    }
}

extern "C" void kernel_launch(void* const* d_in, const int* in_sizes, int n_in,
                              void* d_out, int out_size, void* d_ws, size_t ws_size,
                              hipStream_t stream) {
    (void)in_sizes; (void)n_in; (void)out_size; (void)ws_size;
    const float* x    = (const float*)d_in[0];   // [B,T,IN]
    const float* W    = (const float*)d_in[1];   // [IN+LIQ, LIQ]
    const float* Wro  = (const float*)d_in[2];   // [LIQ, OUT]
    const float* bro  = (const float*)d_in[3];   // [OUT]
    float* y  = (float*)d_out;                   // outputs then h_final
    float* hf = y + (size_t)B_ * T_ * OUT_;

    uint32_t* wxh_pack = (uint32_t*)d_ws;                 // 640 KB
    uint32_t* wro_pack = wxh_pack + (KTOT * LIQ_ / 2);    // 128 KB

    int ndw1 = KTOT * LIQ_ / 2;  // 163840 dwords
    lsm_pack_b<<<(ndw1 + 255) / 256, 256, 0, stream>>>(W, LIQ_, KB_W, wxh_pack, ndw1);
    int ndw2 = LIQ_ * OUT_ / 2;  // 32768 dwords
    lsm_pack_b<<<(ndw2 + 255) / 256, 256, 0, stream>>>(Wro, OUT_, KB_RO, wro_pack, ndw2);

    lsm_main<<<B_ / 16, 256, 0, stream>>>(x, (const uint4*)wxh_pack,
                                          (const uint4*)wro_pack, bro, y, hf);
}
